// ResBlock_3d_1786706395362
// MI455X (gfx1250) — matrix operand order repaired and checked by reading, not compiled, not run on hardware
//
#include <hip/hip_runtime.h>
#include <hip/hip_bf16.h>

typedef __attribute__((ext_vector_type(16))) _Float16 v16h;
typedef __attribute__((ext_vector_type(8)))  float    v8f;

#define Dd 8
#define Hh 48
#define Ww 48
#define Pp (Dd*Hh*Ww)     /* 18432 spatial positions */
#define Cc 32             /* channels */
#define NK 27             /* 3x3x3 taps */
#define MOFF 54           /* offset conv output channels (2*27) */
#define NTt (Pp/16)       /* 1152 position tiles */

// ---------------------------------------------------------------------------
// Pack a weight tensor [M][C][27] into WMMA A-fragment layout:
//   afrag[k][mtile][lane][16 f16]
// A 16x32 f16 layout used here (contiguous k, not ISA table order): lane l (m = l&15, g = l>>4),
//   half j -> K(channel) = g*16 + j
// Runs once; GEMM then loads A with one 32B vector load per chunk.
// ---------------------------------------------------------------------------
__global__ void __launch_bounds__(32)
pack_a_frags(const float* __restrict__ wgt,  // [M][C][27]
             _Float16* __restrict__ afrag,   // [NK][MT][32][16]
             int M, int MT)
{
    const int k    = blockIdx.x;            // tap
    const int mt   = blockIdx.y;            // M tile
    const int lane = threadIdx.x;
    const int g    = lane >> 4;
    const int m    = mt * 16 + (lane & 15);
    v16h vals;
    #pragma unroll
    for (int j = 0; j < 16; ++j) {
        const int v  = j >> 1, hb = j & 1;
        const int c  = g * 16 + j;
        vals[j] = (m < M) ? (_Float16)wgt[((size_t)m * Cc + c) * NK + k]
                          : (_Float16)0.0f;
    }
    *(v16h*)(afrag + (((size_t)k * MT + mt) * 32 + lane) * 16) = vals;
}

// ---------------------------------------------------------------------------
// Regular-conv im2col (zero pad) straight into B-fragment layout:
//   bfrag[k][ptile][lane][16 f16]
// B 32x16 f16 layout: lane l (n = l&15, g = l>>4), half j -> K = g*16 + j.
// One spatial source address per lane per tap; 32B coalesced store.
// ---------------------------------------------------------------------------
__global__ void __launch_bounds__(288)
pack_im2col_frags(const float* __restrict__ in,   // [C][P]
                  _Float16* __restrict__ bfrag)   // [NK][NTt][32][16]
{
    const int t    = blockIdx.x;                       // position tile
    const int k    = blockIdx.y * 9 + threadIdx.y;     // tap 0..26
    const int lane = threadIdx.x;
    const int g    = lane >> 4;
    const int n    = lane & 15;

    const int p   = t * 16 + n;
    const int d   = p / (Hh * Ww);
    const int rem = p % (Hh * Ww);
    const int h   = rem / Ww;
    const int w   = rem % Ww;
    const int kd  = k / 9, kh = (k / 3) % 3, kw = k % 3;
    const int sd  = d + kd - 1, sh = h + kh - 1, sw = w + kw - 1;
    const bool valid = (sd >= 0) & (sd < Dd) & (sh >= 0) & (sh < Hh) &
                       (sw >= 0) & (sw < Ww);
    const int   s  = valid ? (sd * Hh * Ww + sh * Ww + sw) : 0;
    const float sc = valid ? 1.0f : 0.0f;

    v16h vals;
    #pragma unroll
    for (int j = 0; j < 16; ++j) {
        const int c = g * 16 + j;
        vals[j] = (_Float16)(in[(size_t)c * Pp + s] * sc);
    }
    *(v16h*)(bfrag + (((size_t)k * NTt + t) * 32 + lane) * 16) = vals;
}

// ---------------------------------------------------------------------------
// Deformable bilinear H/W sampling -> B-fragment layout f16.
// One thread per (tap k, position p): 4 corner coefs computed once, then a
// 32-channel gather/blend; results written as two contiguous 32B half-groups
// (lane n for c=0..15, lane n+16 for c=16..31).
// ---------------------------------------------------------------------------
__global__ void __launch_bounds__(256)
sample_cols_frag(const float* __restrict__ x,    // [C][P]
                 const float* __restrict__ off,  // [54][P] (2k=h, 2k+1=w)
                 _Float16* __restrict__ bfrag)   // [NK][NTt][32][16]
{
    const int p = blockIdx.x * blockDim.x + threadIdx.x;
    const int k = blockIdx.y;
    if (p >= Pp) return;

    const int kd = k / 9, kh = (k / 3) % 3, kw = k % 3;
    const int d   = p / (Hh * Ww);
    const int rem = p % (Hh * Ww);
    const int h   = rem / Ww;
    const int w   = rem % Ww;

    const int  db = d + kd - 1;
    const bool vd = (db >= 0) && (db < Dd);
    const int  di = min(max(db, 0), Dd - 1);

    const float off_h = off[(size_t)(2 * k)     * Pp + p];
    const float off_w = off[(size_t)(2 * k + 1) * Pp + p];
    const float hf = (float)h + (float)(kh - 1) + off_h;
    const float wf = (float)w + (float)(kw - 1) + off_w;
    const float h0f = floorf(hf), w0f = floorf(wf);
    const float lh = hf - h0f,    lw = wf - w0f;
    const int h0 = (int)h0f, w0 = (int)w0f;
    const int h1 = h0 + 1,   w1 = w0 + 1;

    int   sidx[4];
    float coef[4];
    const int dof = di * Hh * Ww;
    auto mk = [&](int i, int hi, int wi, float wt) {
        const bool v  = vd && (hi >= 0) && (hi < Hh) && (wi >= 0) && (wi < Ww);
        const int  hc = min(max(hi, 0), Hh - 1);
        const int  wc = min(max(wi, 0), Ww - 1);
        sidx[i] = dof + hc * Ww + wc;
        coef[i] = v ? wt : 0.0f;
    };
    mk(0, h0, w0, (1.0f - lh) * (1.0f - lw));
    mk(1, h0, w1, (1.0f - lh) * lw);
    mk(2, h1, w0, lh * (1.0f - lw));
    mk(3, h1, w1, lh * lw);

    v16h lo, hi;
    #pragma unroll
    for (int c = 0; c < Cc; ++c) {
        const float* xc = x + (size_t)c * Pp;
        const float  v  = coef[0] * xc[sidx[0]] + coef[1] * xc[sidx[1]] +
                          coef[2] * xc[sidx[2]] + coef[3] * xc[sidx[3]];
        if (c < 16) lo[c]      = (_Float16)v;
        else        hi[c - 16] = (_Float16)v;
    }
    const int t = p >> 4, n = p & 15;
    _Float16* base = bfrag + ((size_t)(k * NTt + t) * 32) * 16;
    *(v16h*)(base + (size_t)n        * 16) = lo;  // lanes 0-15: K=0..15
    *(v16h*)(base + (size_t)(n + 16) * 16) = hi;  // lanes 16-31: K=16..31
}

// ---------------------------------------------------------------------------
// Unified fragment GEMM: D[M][P] = sum_k A(k) x B(k) + bias (+ residual).
// Inner loop: 2x b128 A-load + 2x b128 B-load + 1 wmma + prefetch of the
// next streamed B fragment (global_prefetch_b8).
// blockDim = (32, MT): one wave per M tile; grid.x = position tiles.
// ---------------------------------------------------------------------------
__global__ void __launch_bounds__(128)
frag_gemm_wmma(const _Float16* __restrict__ afrag,  // [NK][MT][32][16]
               const _Float16* __restrict__ bfrag,  // [NK][NTt][32][16]
               const float* __restrict__ bias,      // [M]
               const float* __restrict__ resid,     // [M][P] or nullptr
               float* __restrict__ outp,            // [M][P]
               int M, int add_resid)
{
    const int lane = threadIdx.x;
    const int mt   = threadIdx.y;
    const int MT   = blockDim.y;
    const int t    = blockIdx.x;
    const int g    = lane >> 4;
    const int n    = lane & 15;

    const size_t a_stride = (size_t)MT  * 32 * 16;
    const size_t b_stride = (size_t)NTt * 32 * 16;
    const _Float16* ap = afrag + ((size_t)mt * 32 + lane) * 16;
    const _Float16* bp = bfrag + ((size_t)t  * 32 + lane) * 16;

    v8f acc = {};
    for (int k = 0; k < NK; ++k) {
        const v16h afr = *(const v16h*)ap;
        const v16h bfr = *(const v16h*)bp;
        ap += a_stride;
        bp += b_stride;
        __builtin_prefetch(bp, 0, 1);   // next streamed B fragment
        acc = __builtin_amdgcn_wmma_f32_16x16x32_f16(
                  false, afr, false, bfr, (short)0, acc, false, false);
    }

    // D layout: lane -> N = n, VGPR r -> M = mt*16 + r + g*8
    #pragma unroll
    for (int r = 0; r < 8; ++r) {
        const int o = mt * 16 + r + g * 8;
        if (o < M) {
            float val = acc[r] + bias[o];
            if (add_resid) val += resid[(size_t)o * Pp + t * 16 + n];
            outp[(size_t)o * Pp + t * 16 + n] = val;
        }
    }
}

// ---------------------------------------------------------------------------
// mish(x) = x * tanh(softplus(x))
// ---------------------------------------------------------------------------
__global__ void __launch_bounds__(256)
mish_kernel(const float* __restrict__ in, float* __restrict__ out, int nElem)
{
    const int i = blockIdx.x * blockDim.x + threadIdx.x;
    if (i < nElem) {
        const float xv = in[i];
        const float sp = (xv > 20.0f) ? xv : log1pf(expf(xv));
        out[i] = xv * tanhf(sp);
    }
}

// ---------------------------------------------------------------------------
extern "C" void kernel_launch(void* const* d_in, const int* in_sizes, int n_in,
                              void* d_out, int out_size, void* d_ws, size_t ws_size,
                              hipStream_t stream)
{
    const float* x      = (const float*)d_in[0];  // [32][18432]
    const float* w_off0 = (const float*)d_in[1];  // [54][32][27]
    const float* b_off0 = (const float*)d_in[2];  // [54]
    const float* w0     = (const float*)d_in[3];  // [32][32][27]
    const float* b0     = (const float*)d_in[4];  // [32]
    const float* w_off1 = (const float*)d_in[5];
    const float* b_off1 = (const float*)d_in[6];
    const float* w1     = (const float*)d_in[7];
    const float* b1     = (const float*)d_in[8];
    float* out = (float*)d_out;

    // ---- workspace partition (all chunks 32B-aligned) ----
    const size_t A_OFF_BYTES  = (size_t)NK * 4 * 32 * 16 * 2;  // 110592 B
    const size_t A_MAIN_BYTES = (size_t)NK * 2 * 32 * 16 * 2;  //  55296 B
    const size_t OFF_BYTES    = (size_t)MOFF * Pp * 4;         // ~4.0 MB
    const size_t BCOLS_BYTES  = (size_t)NK * NTt * 32 * 16 * 2;// ~31.9 MB
    const size_t Y_BYTES      = (size_t)Cc * Pp * 4;           // ~2.4 MB

    char* ws = (char*)d_ws;
    _Float16* a_off0 = (_Float16*)ws;                 ws += A_OFF_BYTES;
    _Float16* a_off1 = (_Float16*)ws;                 ws += A_OFF_BYTES;
    _Float16* a_w0   = (_Float16*)ws;                 ws += A_MAIN_BYTES;
    _Float16* a_w1   = (_Float16*)ws;                 ws += A_MAIN_BYTES;
    float*    off    = (float*)ws;                    ws += OFF_BYTES;
    _Float16* bcols  = (_Float16*)ws;                 ws += BCOLS_BYTES;
    float*    y0     = (float*)ws;                    ws += Y_BYTES;
    float*    m0     = (float*)ws;

    dim3 gPackOff(NK, 4), gPackMain(NK, 2), bPack(32);
    dim3 gIm2col(NTt, 3), bIm2col(32, 9);             // 3*9 = 27 taps
    dim3 gSmp(Pp / 256, NK), bSmp(256);
    dim3 gGemm(NTt);
    dim3 bGemmOff(32, 4), bGemmMain(32, 2);
    dim3 gEw((Cc * Pp + 255) / 256), bEw(256);

    // ---- pack all weight tensors into A-fragment layout (once) ----
    pack_a_frags<<<gPackOff,  bPack, 0, stream>>>(w_off0, a_off0, MOFF, 4);
    pack_a_frags<<<gPackOff,  bPack, 0, stream>>>(w_off1, a_off1, MOFF, 4);
    pack_a_frags<<<gPackMain, bPack, 0, stream>>>(w0,     a_w0,   Cc,   2);
    pack_a_frags<<<gPackMain, bPack, 0, stream>>>(w1,     a_w1,   Cc,   2);

    // ---- layer 0 ----
    pack_im2col_frags<<<gIm2col, bIm2col, 0, stream>>>(x, bcols);
    frag_gemm_wmma<<<gGemm, bGemmOff, 0, stream>>>(a_off0, bcols, b_off0,
                                                   nullptr, off, MOFF, 0);
    sample_cols_frag<<<gSmp, bSmp, 0, stream>>>(x, off, bcols);
    frag_gemm_wmma<<<gGemm, bGemmMain, 0, stream>>>(a_w0, bcols, b0,
                                                    nullptr, y0, Cc, 0);
    mish_kernel<<<gEw, bEw, 0, stream>>>(y0, m0, Cc * Pp);

    // ---- layer 1 (+ residual fused into final store) ----
    pack_im2col_frags<<<gIm2col, bIm2col, 0, stream>>>(m0, bcols);
    frag_gemm_wmma<<<gGemm, bGemmOff, 0, stream>>>(a_off1, bcols, b_off1,
                                                   nullptr, off, MOFF, 0);
    sample_cols_frag<<<gSmp, bSmp, 0, stream>>>(m0, off, bcols);
    frag_gemm_wmma<<<gGemm, bGemmMain, 0, stream>>>(a_w1, bcols, b1,
                                                    x, out, Cc, 1);
}
